// IntraAgg_75015898792523
// MI455X (gfx1250) — matrix operand order repaired and checked by reading, not compiled, not run on hardware
//
#include <hip/hip_runtime.h>
#include <hip/hip_bf16.h>
#include <stdint.h>

// ---------------------------------------------------------------------------
// Types for CDNA5 WMMA (wave32): D(16x16 f32) = A(16x32 bf16) x B(32x16 bf16) + C
// ---------------------------------------------------------------------------
typedef __bf16 bf16_t;
typedef __bf16 v16bf __attribute__((ext_vector_type(16)));
typedef __bf16 v8bf  __attribute__((ext_vector_type(8)));
typedef float  v8f   __attribute__((ext_vector_type(8)));

// gfx1250 async global->LDS copy (ASYNCcnt-tracked), if the toolchain exposes it
#if defined(__has_builtin)
#if __has_builtin(__builtin_amdgcn_global_load_async_to_lds_b128)
#define USE_ASYNC_COPY 1
#endif
#endif
#ifndef USE_ASYNC_COPY
#define USE_ASYNC_COPY 0
#endif

#if USE_ASYNC_COPY
typedef int v4i __attribute__((vector_size(16)));
typedef __attribute__((address_space(1))) v4i as1_v4i;   // global pointee
typedef __attribute__((address_space(3))) v4i as3_v4i;   // LDS pointee
__device__ __forceinline__ void async_copy16(const void* g, void* l) {
    // copies 16B per lane: LDS[l] = MEM[g]; tracked by ASYNCcnt
    __builtin_amdgcn_global_load_async_to_lds_b128(
        (as1_v4i*)(uintptr_t)g, (as3_v4i*)(uint32_t)(uintptr_t)l, 0, 0);
}
template <int N>
__device__ __forceinline__ void wait_async() {
#if __has_builtin(__builtin_amdgcn_s_wait_asynccnt)
    __builtin_amdgcn_s_wait_asynccnt(N);
#else
    if (N == 0) asm volatile("s_wait_asynccnt 0x0" ::: "memory");
    else        asm volatile("s_wait_asynccnt 0x8" ::: "memory");
#endif
}
#endif

__device__ __forceinline__ bf16_t f2bf(float f) {
    // round-to-nearest-even f32 -> bf16
    union { float f; uint32_t u; } a; a.f = f;
    uint32_t r = a.u + 0x7FFFu + ((a.u >> 16) & 1u);
    uint16_t h = (uint16_t)(r >> 16);
    return __builtin_bit_cast(bf16_t, h);
}

// ---------------------------------------------------------------------------
// K0: convert f32 weight [K][N] -> bf16 transposed [N][K]
// ---------------------------------------------------------------------------
__global__ __launch_bounds__(256) void k_convT(const float* __restrict__ src,
                                               bf16_t* __restrict__ dst,
                                               int K, int N) {
    int t = blockIdx.x * 256 + threadIdx.x;
    if (t >= K * N) return;
    int k = t / N;
    int n = t - k * N;
    dst[(size_t)n * K + k] = f2bf(src[t]);
}

// ---------------------------------------------------------------------------
// K1: gather + mean pool.  block = one seed node, 128 threads x float4 = 512 feats
// ---------------------------------------------------------------------------
__global__ __launch_bounds__(128) void k_gather(const float* __restrict__ feat,
                                                const int* __restrict__ neigh,
                                                float* __restrict__ aggOut,
                                                bf16_t* __restrict__ catA,
                                                bf16_t* __restrict__ envA) {
    const int b = blockIdx.x;
    const int t = threadIdx.x;                 // float4 slot 0..127
    const int* nb = neigh + (size_t)b * 32;

    float sx = 0.f, sy = 0.f, sz = 0.f, sw = 0.f;
    float ex = 0.f, ey = 0.f, ez = 0.f, ew = 0.f; // self (neigh[0] == node)
    for (int k = 0; k < 32; ++k) {
        int idx = nb[k];
        const float4 v = ((const float4*)(feat + (size_t)idx * 512))[t];
        sx += v.x; sy += v.y; sz += v.z; sw += v.w;
        if (k == 0) { ex = v.x; ey = v.y; ez = v.z; ew = v.w; }
    }
    const float inv32 = 1.f / 32.f, inv31 = 1.f / 31.f;
    float ax = sx * inv32, ay = sy * inv32, az = sz * inv32, aw = sw * inv32;
    float vx = (sx - ex) * inv31, vy = (sy - ey) * inv31,
          vz = (sz - ez) * inv31, vw = (sw - ew) * inv31;

    float4 aggv; aggv.x = ax; aggv.y = ay; aggv.z = az; aggv.w = aw;
    ((float4*)(aggOut + (size_t)b * 512))[t] = aggv;

    bf16_t* catRow = catA + (size_t)b * 1024;
    int c = t * 4;
    catRow[c + 0] = f2bf(ex); catRow[c + 1] = f2bf(ey);
    catRow[c + 2] = f2bf(ez); catRow[c + 3] = f2bf(ew);
    catRow[512 + c + 0] = f2bf(ax); catRow[512 + c + 1] = f2bf(ay);
    catRow[512 + c + 2] = f2bf(az); catRow[512 + c + 3] = f2bf(aw);
    bf16_t* envRow = envA + (size_t)b * 512;
    envRow[c + 0] = f2bf(vx); envRow[c + 1] = f2bf(vy);
    envRow[c + 2] = f2bf(vz); envRow[c + 3] = f2bf(vw);
}

// ---------------------------------------------------------------------------
// K2: bf16 WMMA GEMM.  C[M,N] = A[M,K] * Bt[N,K]^T.
// Block: 256 threads = 8 waves, 128x128 C tile; wave = 32x64 (2x4 accumulators).
// Stage: BK=64 (two v_wmma k-steps). Async double-buffered LDS on gfx1250.
// M,N multiples of 128, K of 64 (>=512). rawOut (template) pre-act, actOut relu.
// ---------------------------------------------------------------------------
#define LDS_ROW_H 72   // halves per LDS tile row: 64 data + 8 pad (144 B, 16B-aligned)

template <bool HAS_RAW>
__global__ __launch_bounds__(256) void k_gemm_bf16(const bf16_t* __restrict__ A,
                                                   const bf16_t* __restrict__ Bt,
                                                   float* __restrict__ rawOut,
                                                   float* __restrict__ actOut,
                                                   int lda, int ldb, int N, int K) {
#if USE_ASYNC_COPY
    __shared__ __align__(16) bf16_t aT[2][128 * LDS_ROW_H];
    __shared__ __align__(16) bf16_t bT[2][128 * LDS_ROW_H];
#else
    __shared__ __align__(16) bf16_t aT[1][128 * LDS_ROW_H];
    __shared__ __align__(16) bf16_t bT[1][128 * LDS_ROW_H];
#endif

    const int tid   = threadIdx.x;
    const int lane  = tid & 31;
    const int wid   = tid >> 5;
    const int waveM = wid >> 1;            // 0..3  -> 32 rows each
    const int waveN = wid & 1;             // 0..1  -> 64 cols each
    const int blockM = blockIdx.x * 128;
    const int blockN = blockIdx.y * 128;
    const int lh   = lane & 15;
    const int hsel = lane >> 4;            // 0: low half-wave, 1: high half-wave

    v8f acc[2][4];
#pragma unroll
    for (int i = 0; i < 2; ++i)
#pragma unroll
        for (int j = 0; j < 4; ++j)
            acc[i][j] = (v8f){0.f, 0.f, 0.f, 0.f, 0.f, 0.f, 0.f, 0.f};

    // ---- fragment compute for one resident stage (BK=64 = 2 wmma k-steps) ----
    auto computeTile = [&](const bf16_t* aBuf, const bf16_t* bBuf) {
#pragma unroll
        for (int kk = 0; kk < 2; ++kk) {
            v16bf aF[2], bF[4];
#pragma unroll
            for (int i = 0; i < 2; ++i) {
                // A: lane<16 holds K{0..7,16..23}; lane>=16 holds K{8..15,24..31}
                const bf16_t* base = aBuf + (waveM * 32 + i * 16 + lh) * LDS_ROW_H
                                          + kk * 32 + hsel * 8;
                v8bf lo = *(const v8bf*)(base);
                v8bf hi = *(const v8bf*)(base + 16);
                aF[i] = __builtin_shufflevector(lo, hi, 0, 1, 2, 3, 4, 5, 6, 7,
                                                8, 9, 10, 11, 12, 13, 14, 15);
            }
#pragma unroll
            for (int j = 0; j < 4; ++j) {
                // B (Bt rows = columns of B): lane<16 K{0..7,16..23}; lane>=16 K{8..15,24..31}
                const bf16_t* base = bBuf + (waveN * 64 + j * 16 + lh) * LDS_ROW_H
                                          + kk * 32 + hsel * 8;
                v8bf lo = *(const v8bf*)(base);
                v8bf hi = *(const v8bf*)(base + 16);
                bF[j] = __builtin_shufflevector(lo, hi, 0, 1, 2, 3, 4, 5, 6, 7,
                                                8, 9, 10, 11, 12, 13, 14, 15);
            }
#pragma unroll
            for (int i = 0; i < 2; ++i)
#pragma unroll
                for (int j = 0; j < 4; ++j)
                    acc[i][j] = __builtin_amdgcn_wmma_f32_16x16x32_bf16(
                        false, aF[i], false, bF[j], (short)0, acc[i][j], false, false);
        }
    };

#if USE_ASYNC_COPY
    // ---- async double-buffered pipeline: 8 async-copy instructions / stage ----
    const int nStages = K >> 6;            // always >= 8 here
    auto issueStage = [&](int s) {
        const int k0  = s << 6;
        const int buf = s & 1;
#pragma unroll
        for (int i = 0; i < 4; ++i) {
            int chunk = tid + i * 256;     // 0..1023 (16B units, 8 per 128B row)
            int row   = chunk >> 3;
            int c16   = chunk & 7;
            async_copy16(A  + (size_t)(blockM + row) * lda + k0 + c16 * 8,
                         &aT[buf][row * LDS_ROW_H + c16 * 8]);
            async_copy16(Bt + (size_t)(blockN + row) * ldb + k0 + c16 * 8,
                         &bT[buf][row * LDS_ROW_H + c16 * 8]);
        }
    };
    issueStage(0);
    int s = 0;
    for (; s < nStages - 1; ++s) {         // steady state: branch-free body
        issueStage(s + 1);                 // fills the other buffer
        wait_async<8>();                   // stage-s ops retired; 8 newest pending
        __syncthreads();                   // stage-s tiles visible to all waves
        computeTile(aT[s & 1], bT[s & 1]);
        __syncthreads();                   // all readers done before buf refill
    }
    wait_async<0>();                       // tail stage
    __syncthreads();
    computeTile(aT[s & 1], bT[s & 1]);
#else
    // ---- fallback: synchronous VGPR-staged copy, single buffer ----
    for (int k0 = 0; k0 < K; k0 += 64) {
#pragma unroll
        for (int i = 0; i < 4; ++i) {
            int chunk = tid + i * 256;
            int row   = chunk >> 3;
            int c16   = chunk & 7;
            const uint4* gA = (const uint4*)(A  + (size_t)(blockM + row) * lda + k0) + c16;
            *(uint4*)(&aT[0][row * LDS_ROW_H + c16 * 8]) = *gA;
            const uint4* gB = (const uint4*)(Bt + (size_t)(blockN + row) * ldb + k0) + c16;
            *(uint4*)(&bT[0][row * LDS_ROW_H + c16 * 8]) = *gB;
        }
        __syncthreads();
        computeTile(aT[0], bT[0]);
        __syncthreads();
    }
#endif

    // ---- epilogue: C layout -> VGPR r: lanes0-15 M=r, lanes16-31 M=r+8 ----
    const int rowBase = blockM + waveM * 32;
    const int colBase = blockN + waveN * 64;
#pragma unroll
    for (int i = 0; i < 2; ++i) {
#pragma unroll
        for (int r = 0; r < 8; ++r) {
            const size_t row  = (size_t)(rowBase + i * 16 + hsel * 8 + r);
            const size_t base = row * (size_t)N + (size_t)(colBase + lh);
#pragma unroll
            for (int j = 0; j < 4; ++j) {
                float v = acc[i][j][r];
                if (HAS_RAW) rawOut[base + j * 16] = v;
                actOut[base + j * 16] = v > 0.f ? v : 0.f;
            }
        }
    }
}

// ---------------------------------------------------------------------------
// K3: batchnorm statistics + fold into scale/shift; build bf16 BN'ed A matrix
// feats[b, c] = (c<512 ? agg[b,c] : gen[b,c-512]),  B=8192, C=1536
// ---------------------------------------------------------------------------
__global__ __launch_bounds__(256) void k_zero(float* p, int n) {
    int i = blockIdx.x * 256 + threadIdx.x;
    if (i < n) p[i] = 0.f;
}

__global__ __launch_bounds__(256) void k_bnstats(const float* __restrict__ agg,
                                                 const float* __restrict__ gen,
                                                 float* __restrict__ sums,
                                                 float* __restrict__ sumsq) {
    int c = blockIdx.x * 256 + threadIdx.x;   // 0..1535 (grid.x = 6)
    int r0 = blockIdx.y * 256;                // grid.y = 32 row slices
    float s = 0.f, s2 = 0.f;
    if (c < 512) {
        for (int r = r0; r < r0 + 256; ++r) {
            float x = agg[(size_t)r * 512 + c];
            s += x; s2 += x * x;
        }
    } else {
        int cc = c - 512;
        for (int r = r0; r < r0 + 256; ++r) {
            float x = gen[(size_t)r * 1024 + cc];
            s += x; s2 += x * x;
        }
    }
    atomicAdd(&sums[c], s);
    atomicAdd(&sumsq[c], s2);
}

__global__ __launch_bounds__(256) void k_bnfinal(const float* __restrict__ sums,
                                                 const float* __restrict__ sumsq,
                                                 const float* __restrict__ gamma,
                                                 const float* __restrict__ beta,
                                                 float* __restrict__ scale,
                                                 float* __restrict__ shift) {
    int c = blockIdx.x * 256 + threadIdx.x;
    if (c >= 1536) return;
    const float invB = 1.f / 8192.f;
    float mean = sums[c] * invB;
    float var  = sumsq[c] * invB - mean * mean;  // biased, as jnp.var
    float sc   = gamma[c] * rsqrtf(var + 1e-5f);
    scale[c] = sc;
    shift[c] = beta[c] - mean * sc;
}

__global__ __launch_bounds__(256) void k_abn(const float* __restrict__ agg,
                                             const float* __restrict__ gen,
                                             const float* __restrict__ scale,
                                             const float* __restrict__ shift,
                                             bf16_t* __restrict__ Abn) {
    int c = blockIdx.x * 256 + threadIdx.x;   // 0..1535
    int b = blockIdx.y;                       // 0..8191
    float x = (c < 512) ? agg[(size_t)b * 512 + c]
                        : gen[(size_t)b * 1024 + (c - 512)];
    Abn[(size_t)b * 1536 + c] = f2bf(x * scale[c] + shift[c]);
}

// ---------------------------------------------------------------------------
// Host-side orchestration
// ---------------------------------------------------------------------------
extern "C" void kernel_launch(void* const* d_in, const int* in_sizes, int n_in,
                              void* d_out, int out_size, void* d_ws, size_t ws_size,
                              hipStream_t stream) {
    (void)in_sizes; (void)n_in; (void)out_size; (void)ws_size;

    const float* feat  = (const float*)d_in[0];   // [100000, 512]
    const float* Wgen  = (const float*)d_in[1];   // [1024, 1024]
    const float* W1    = (const float*)d_in[2];   // [1536, 1536]
    const float* gamma = (const float*)d_in[3];   // [1536]
    const float* beta  = (const float*)d_in[4];   // [1536]
    const int*   neigh = (const int*)d_in[6];     // [8192, 32] (col0 == nodes)

    float* out = (float*)d_out;                   // tuple, flat in return order
    float* out_agg  = out;                        // [8192, 512]
    float* out_to   = out + 4194304;              // [8192, 1536]
    float* out_gen  = out + 16777216;             // [8192, 1024]
    float* out_raw  = out + 25165824;             // [8192, 1024]
    float* out_egen = out + 33554432;             // [8192, 1024]
    float* out_eraw = out + 41943040;             // [8192, 1024]

    char* ws = (char*)d_ws;
    bf16_t* WgenT = (bf16_t*)(ws);                // [1024][1024] bf16 (N-major)
    bf16_t* W1T   = (bf16_t*)(ws + 2097152);      // [1536][1536] bf16 (N-major)
    bf16_t* catA  = (bf16_t*)(ws + 6815744);      // [8192][1024] bf16
    bf16_t* envA  = (bf16_t*)(ws + 23592960);     // [8192][512]  bf16
    bf16_t* Abn   = (bf16_t*)(ws + 31981568);     // [8192][1536] bf16
    float*  sums  = (float*)(ws + 57147392);      // [1536]
    float*  sumsq = sums + 1536;                  // [1536]
    float*  scale = sumsq + 1536;                 // [1536]
    float*  shift = scale + 1536;                 // [1536]

    // 0) weights -> bf16, transposed [N][K]
    k_convT<<<(1024 * 1024 + 255) / 256, 256, 0, stream>>>(Wgen, WgenT, 1024, 1024);
    k_convT<<<(1536 * 1536 + 255) / 256, 256, 0, stream>>>(W1, W1T, 1536, 1536);

    // 1) gather + pool: agg (out0), cat = [self||agg] bf16, env bf16
    k_gather<<<8192, 128, 0, stream>>>(feat, neigh, out_agg, catA, envA);

    // 2) raw = cat @ Wgen (out3), gen = relu(raw) (out2)
    k_gemm_bf16<true><<<dim3(64, 8), 256, 0, stream>>>(catA, WgenT, out_raw, out_gen,
                                                       1024, 1024, 1024, 1024);
    // 3) env_raw = env @ Wgen[512:,:] (out5), env_gen = relu (out4)
    k_gemm_bf16<true><<<dim3(64, 8), 256, 0, stream>>>(envA, WgenT + 512, out_eraw,
                                                       out_egen, 512, 1024, 1024, 512);

    // 4) batchnorm stats over feats = [agg || gen], fold to scale/shift
    k_zero<<<(3072 + 255) / 256, 256, 0, stream>>>(sums, 3072);
    k_bnstats<<<dim3(6, 32), 256, 0, stream>>>(out_agg, out_gen, sums, sumsq);
    k_bnfinal<<<6, 256, 0, stream>>>(sums, sumsq, gamma, beta, scale, shift);
    k_abn<<<dim3(6, 8192), 256, 0, stream>>>(out_agg, out_gen, scale, shift, Abn);

    // 5) to_feats = relu(BN(feats) @ weight1) (out1)
    k_gemm_bf16<false><<<dim3(64, 12), 256, 0, stream>>>(Abn, W1T, nullptr, out_to,
                                                         1536, 1536, 1536, 1536);
}
